// RGCNTransformerModelWithHourlyHeads_31164282699781
// MI455X (gfx1250) — matrix operand order repaired and checked by reading, not compiled, not run on hardware
//
#include <hip/hip_runtime.h>
#include <hip/hip_bf16.h>

// ---------------------------------------------------------------------------
// RGCN + Transformer forward for MI455X (gfx1250).
//
// Roofline: ~100 GFLOP of dense GEMMs over 8000/96000-row activations that
// fit in the 192MB L2 -> matrix-pipe bound. All big GEMMs (M%32==0, N%32==0,
// K%32==0) run a 32x128-tile WMMA kernel: operands staged to LDS with
// GLOBAL_LOAD_ASYNC_TO_LDS_B128 (ASYNCcnt), consumed by
// v_wmma_f32_16x16x32_bf16 with f32 accumulate (bf16 convert via native
// v_cvt; weights are 0.02-scaled so bf16 inputs are numerically adequate).
// Ragged small GEMMs (152/140/16/7 wide) use a bounds-checked fallback.
// Irregular RGCN message passing = atomic scatter (bandwidth-bound, cheap).
//
// Input flattening assumption: jax.tree_util.tree_leaves(setup_inputs()),
// i.e. dicts in sorted-key order, lists in order.  Index map:
//   0 batch_assign[8000] i32      1 edge_index[2,64000] i32 (src,dst)
//   2 edge_type[64000] i32        3 env_features[13,40,16] f32
//   4..29  dec_layers[0]:  W1,W2,b1,b2, cross{Wk,Wo,Wq,Wv,bk,bo,bq,bv},
//                          n1b,n1g,n2b,n2g,n3b,n3g, self{Wk..bv}
//   30..55 dec_layers[1]
//   56..71 enc_layers[0]:  W1,W2, attn{Wk,Wo,Wq,Wv,bk,bo,bq,bv}, b1,b2,
//                          n1b,n1g,n2b,n2g
//   72..87 enc_layers[1]
//   88..93 env_enc: W1,W2,b1,b2,be1,g1
//   94..99 fusion:  W1,W2,b1,b2,be1,g1
//   100 head_W1[12,128,64]  101 head_W2[12,64,1]
//   102 head_b1[12,64]      103 head_b2[12,1]
//   104..121 rgcn_dec L0..L2: Wrel,Wroot,alpha,bias,bn_b,bn_g
//   122..139 rgcn_enc L0..L2
//   140..145 time_enc: W1,W2,b1,b2,be1,g1
//   146 timeline_time_features[12,6]   147 x_all[13,8000,32]
// ---------------------------------------------------------------------------

#define NN 8000
#define NE 64000
#define NG 40
#define NREL 5
#define TP 12
#define ND (TP * NN)      // 96000 decoder rows
#define DM 128
#define DFF 512
#define FIN 152           // fusion concat width
#define FHID 140

typedef __bf16 v16bf __attribute__((ext_vector_type(16)));
typedef float  v8f   __attribute__((ext_vector_type(8)));

union BF16x16 { v16bf v; __bf16 e[16]; };

__device__ inline __bf16 bfc(float x) { return (__bf16)x; }  // native cvt (RNE)

__device__ inline float wave_sum(float v) {
#pragma unroll
  for (int off = 16; off > 0; off >>= 1) v += __shfl_xor(v, off, 32);
  return v;
}

#define GF_RELU 1
#define GF_ACC  2

// ---------------------------------------------------------------------------
// FAST GEMM (M%32==0, N%32==0, K%32==0): 128 threads = 4 waves.
// Block tile 32(M) x 128(N); wave w owns columns [w*32, w*32+32).
// Per K-step: A slab 32x32 f32 (4KB) + B slab 32x128 f32 (16KB) are staged
// into LDS via global_load_async_to_lds_b128 (ASYNCcnt), then each wave does
// 4x v_wmma_f32_16x16x32_bf16.  No predication around the WMMAs.
// ---------------------------------------------------------------------------
__global__ __launch_bounds__(128) void wmma_gemm_fast_k(
    const float* __restrict__ A, const float* __restrict__ W,
    const float* __restrict__ bias, float* __restrict__ C,
    int M, int N, int K, int flags) {
  __shared__ float As[32 * 32];     // [m][k]
  __shared__ float Bs[32 * 128];    // [k][n]
  const int tid  = threadIdx.x;
  const int wv   = tid >> 5;
  const int lane = tid & 31;
  const int half = lane >> 4;       // 0/1
  const int l15  = lane & 15;
  const int row0 = blockIdx.y * 32;
  const int colB = blockIdx.x * 128;          // block N origin
  const int colW = colB + wv * 32;            // wave N origin
  const bool waveActive = colW < N;           // wave-uniform

  v8f acc[2][2];
#pragma unroll
  for (int mt = 0; mt < 2; ++mt)
#pragma unroll
    for (int nt = 0; nt < 2; ++nt) {
      v8f z = {0.f, 0.f, 0.f, 0.f, 0.f, 0.f, 0.f, 0.f};
      acc[mt][nt] = z;
    }
  if (waveActive && (flags & GF_ACC)) {
#pragma unroll
    for (int mt = 0; mt < 2; ++mt)
#pragma unroll
      for (int nt = 0; nt < 2; ++nt)
#pragma unroll
        for (int v = 0; v < 8; ++v)
          acc[mt][nt][v] =
              C[(size_t)(row0 + mt * 16 + v + half * 8) * N + (colW + nt * 16 + l15)];
  }

  // staging role: thread -> (slab row, quarter)
  const int srow = tid >> 2;        // 0..31
  const int sq   = tid & 3;         // 0..3
  const unsigned ldsA  = (unsigned)(unsigned long long)(const void*)&As[srow * 32 + sq * 8];
  const unsigned ldsB0 = (unsigned)(unsigned long long)(const void*)&Bs[srow * 128 + sq * 32];

  for (int k0 = 0; k0 < K; k0 += 32) {
    __syncthreads();  // protect LDS reuse across iterations
    // A: row srow, k-range [k0+sq*8, k0+sq*8+8) -> two b128 async loads
    {
      unsigned long long ga =
          (unsigned long long)(const void*)(A + (size_t)(row0 + srow) * K + k0 + sq * 8);
      asm volatile("global_load_async_to_lds_b128 %0, %1, off"
                   :: "v"(ldsA), "v"(ga) : "memory");
      asm volatile("global_load_async_to_lds_b128 %0, %1, off"
                   :: "v"(ldsA + 16u), "v"(ga + 16ull) : "memory");
    }
    // B: k-row srow, columns colB+sq*32 .. +32 -> eight b128 async loads
#pragma unroll
    for (int j = 0; j < 8; ++j) {
      int c = sq * 32 + j * 4;
      if (colB + c < N) {
        unsigned long long gb =
            (unsigned long long)(const void*)(W + (size_t)(k0 + srow) * N + colB + c);
        asm volatile("global_load_async_to_lds_b128 %0, %1, off"
                     :: "v"(ldsB0 + (unsigned)(j * 16)), "v"(gb) : "memory");
      }
    }
    asm volatile("s_wait_asynccnt 0x0" ::: "memory");
    __syncthreads();

    if (waveActive) {
      BF16x16 a[2], b[2];
      // A tiles: lanes 0-15 -> K {0..7,16..23}; lanes 16-31 -> {8..15,24..31}
#pragma unroll
      for (int mt = 0; mt < 2; ++mt) {
        const int r = mt * 16 + l15;
#pragma unroll
        for (int i = 0; i < 8; ++i) {
          a[mt].e[i]     = bfc(As[r * 32 + half * 8 + i]);
          a[mt].e[i + 8] = bfc(As[r * 32 + 16 + half * 8 + i]);
        }
      }
      // B tiles: lane = column; lanes 0-15 -> K {0..7,16..23}; lanes 16-31 -> {8..15,24..31}
#pragma unroll
      for (int nt = 0; nt < 2; ++nt) {
        const int c = wv * 32 + nt * 16 + l15;
#pragma unroll
        for (int i = 0; i < 8; ++i) {
          b[nt].e[i]     = bfc(Bs[(half * 8 + i) * 128 + c]);
          b[nt].e[i + 8] = bfc(Bs[(16 + half * 8 + i) * 128 + c]);
        }
      }
#pragma unroll
      for (int mt = 0; mt < 2; ++mt)
#pragma unroll
        for (int nt = 0; nt < 2; ++nt)
          acc[mt][nt] = __builtin_amdgcn_wmma_f32_16x16x32_bf16(
              false, a[mt].v, false, b[nt].v, (short)0, acc[mt][nt], false, false);
    }
  }

  if (waveActive) {
#pragma unroll
    for (int mt = 0; mt < 2; ++mt)
#pragma unroll
      for (int nt = 0; nt < 2; ++nt)
#pragma unroll
        for (int v = 0; v < 8; ++v) {
          int r = row0 + mt * 16 + v + half * 8;
          int c = colW + nt * 16 + l15;
          float x = acc[mt][nt][v];
          if (bias) x += bias[c];
          if (flags & GF_RELU) x = fmaxf(x, 0.f);
          C[(size_t)r * N + c] = x;
        }
  }
}

// ---------------------------------------------------------------------------
// Generic bounds-checked GEMM (small ragged shapes only). One wave per tile.
// ---------------------------------------------------------------------------
__global__ void wmma_gemm_k(const float* __restrict__ A, const float* __restrict__ W,
                            const float* __restrict__ bias, float* __restrict__ C,
                            int M, int N, int K, int flags) {
  const int lane = threadIdx.x & 31;
  const int col0 = blockIdx.x * 16;
  const int row0 = blockIdx.y * 16;
  const int half = lane >> 4;
  const int l15  = lane & 15;
  const int arow = row0 + l15;
  const int bcol = col0 + l15;
  const bool arow_ok = arow < M;
  const bool bcol_ok = bcol < N;

  v8f acc = {0.f, 0.f, 0.f, 0.f, 0.f, 0.f, 0.f, 0.f};
  if (flags & GF_ACC) {
#pragma unroll
    for (int v = 0; v < 8; ++v) {
      int r = row0 + v + half * 8, c = col0 + l15;
      acc[v] = (r < M && c < N) ? C[(size_t)r * N + c] : 0.f;
    }
  }
  for (int k0 = 0; k0 < K; k0 += 32) {
    if (arow_ok && k0 + 32 < K)
      __builtin_prefetch(&A[(size_t)arow * K + k0 + 32], 0, 1);  // global_prefetch_b8
    BF16x16 a, b;
#pragma unroll
    for (int i = 0; i < 8; ++i) {
      int ka = k0 + half * 8 + i;
      int kb = k0 + 16 + half * 8 + i;
      a.e[i]     = bfc((arow_ok && ka < K) ? A[(size_t)arow * K + ka] : 0.f);
      a.e[i + 8] = bfc((arow_ok && kb < K) ? A[(size_t)arow * K + kb] : 0.f);
    }
#pragma unroll
    for (int i = 0; i < 8; ++i) {
      int ka = k0 + half * 8 + i;
      int kb = k0 + 16 + half * 8 + i;
      b.e[i]     = bfc((bcol_ok && ka < K) ? W[(size_t)ka * N + bcol] : 0.f);
      b.e[i + 8] = bfc((bcol_ok && kb < K) ? W[(size_t)kb * N + bcol] : 0.f);
    }
    acc = __builtin_amdgcn_wmma_f32_16x16x32_bf16(false, a.v, false, b.v,
                                                  (short)0, acc, false, false);
  }
#pragma unroll
  for (int v = 0; v < 8; ++v) {
    int r = row0 + v + half * 8, c = col0 + l15;
    if (r < M && c < N) {
      float x = acc[v];
      if (bias) x += bias[c];
      if (flags & GF_RELU) x = fmaxf(x, 0.f);
      C[(size_t)r * N + c] = x;
    }
  }
}

// ---------------------------------------------------------------------------
// LayerNorm over last dim D with optional residual: Y = LN(X + Res)*g + b.
// ---------------------------------------------------------------------------
__global__ void ln_res_k(const float* __restrict__ X, const float* __restrict__ Res,
                         const float* __restrict__ g, const float* __restrict__ b,
                         float* __restrict__ Y, int M, int D) {
  int row = blockIdx.x;
  if (row >= M) return;
  int lane = threadIdx.x & 31;
  float buf[16];
  int n = 0;
  float s = 0.f;
  for (int j = lane; j < D; j += 32) {
    float v = X[(size_t)row * D + j] + (Res ? Res[(size_t)row * D + j] : 0.f);
    buf[n++] = v; s += v;
  }
  float mean = wave_sum(s) / (float)D;
  float vs = 0.f;
  for (int i = 0; i < n; ++i) { float d = buf[i] - mean; vs += d * d; }
  float inv = rsqrtf(wave_sum(vs) / (float)D + 1e-5f);
  n = 0;
  for (int j = lane; j < D; j += 32)
    Y[(size_t)row * D + j] = (buf[n++] - mean) * inv * g[j] + b[j];
}

// BatchNorm(eval, running 0/1) + PReLU
__global__ void bn_prelu_k(const float* __restrict__ x, const float* __restrict__ bn_g,
                           const float* __restrict__ bn_b, const float* __restrict__ alpha,
                           float* __restrict__ y, size_t n, int D) {
  size_t i = (size_t)blockIdx.x * blockDim.x + threadIdx.x;
  if (i >= n) return;
  int c = (int)(i % D);
  const float BNS = 0.9999950000374997f;  // 1/sqrt(1+1e-5)
  float v = x[i] * (bn_g[c] * BNS) + bn_b[c];
  y[i] = v > 0.f ? v : alpha[c] * v;
}

__global__ void fill_k(float* p, float v, size_t n) {
  size_t i = (size_t)blockIdx.x * blockDim.x + threadIdx.x;
  if (i < n) p[i] = v;
}

__global__ void scale_copy_k(const float* __restrict__ x, float* __restrict__ y,
                             float s, size_t n) {
  size_t i = (size_t)blockIdx.x * blockDim.x + threadIdx.x;
  if (i < n) y[i] = x[i] * s;
}

__global__ void edge_cnt_k(const int* __restrict__ dst, const int* __restrict__ et,
                           float* __restrict__ cnt, int E) {
  int i = blockIdx.x * blockDim.x + threadIdx.x;
  if (i < E) atomicAdd(&cnt[dst[i] * NREL + et[i]], 1.f);
}
__global__ void inv_cnt_k(const float* __restrict__ c, float* __restrict__ inv, int n) {
  int i = blockIdx.x * blockDim.x + threadIdx.x;
  if (i < n) inv[i] = 1.f / fmaxf(c[i], 1.f);
}

// scatter mean: agg[s,dst] += xw[s,src] / cnt[dst,rel] for edges of relation `rel`
__global__ void rgcn_scatter_k(const float* __restrict__ xw, const int* __restrict__ src,
                               const int* __restrict__ dst, const int* __restrict__ et,
                               const float* __restrict__ inv, float* __restrict__ agg,
                               int rel, int E, int S, int D) {
  int t = blockIdx.x * blockDim.x + threadIdx.x;
  if (t >= E * S) return;
  int e = t % E, s = t / E;
  if (et[e] != rel) return;
  int sn = src[e], dn = dst[e];
  float w = inv[dn * NREL + rel];
  const float* mp = xw + ((size_t)s * NN + sn) * D;
  float* op = agg + ((size_t)s * NN + dn) * D;
  for (int c = 0; c < D; ++c) atomicAdd(&op[c], mp[c] * w);
}

// fusion input concat: [h(128) | env_emb[batch](16) | time_emb(8)]
__global__ void fusion_concat_k(const float* __restrict__ h, const float* __restrict__ env,
                                const float* __restrict__ temb, const int* __restrict__ batch,
                                float* __restrict__ out, int S, int env_t0, int temb_stride) {
  size_t i = (size_t)blockIdx.x * blockDim.x + threadIdx.x;
  size_t tot = (size_t)S * NN * FIN;
  if (i >= tot) return;
  int c = (int)(i % FIN);
  size_t rn = i / FIN;
  int n = (int)(rn % NN);
  int s = (int)(rn / NN);
  float v;
  if (c < 128)      v = h[rn * 128 + c];
  else if (c < 144) v = env[((size_t)(env_t0 + s) * NG + batch[n]) * 16 + (c - 128)];
  else              v = temb[s * temb_stride + (c - 144)];
  out[i] = v;
}

__global__ void compute_pe_k(float* pe, int S) {
  int i = blockIdx.x * blockDim.x + threadIdx.x;
  if (i >= S * DM) return;
  int t = i / DM, d = i % DM, pair = d >> 1;
  float div = __expf(-(float)(2 * pair) * 9.210340371976184f / 128.0f);
  float ang = (float)t * div;
  pe[i] = (d & 1) ? __cosf(ang) : __sinf(ang);
}

__global__ void add_pe_k(float* __restrict__ x, const float* __restrict__ pe,
                         size_t rowsPerStep, size_t n) {
  size_t i = (size_t)blockIdx.x * blockDim.x + threadIdx.x;
  if (i >= n) return;
  int d = (int)(i % DM);
  int s = (int)(i / (rowsPerStep * DM));
  x[i] += pe[s * DM + d];
}

// attention: wave per (s,b,h); dh=32 => one lane per channel (wave32-native)
__global__ void mha_attn_k(const float* __restrict__ Q, const float* __restrict__ Kk,
                           const float* __restrict__ V, float* __restrict__ O,
                           int S, int T, int B, int causal) {
  int wid = blockIdx.x;               // ((s*B)+b)*4 + h
  int lane = threadIdx.x & 31;
  int h = wid & 3;
  int sb = wid >> 2;
  int b = sb % B, s = sb / B;
  float q = Q[((size_t)s * B + b) * DM + h * 32 + lane];
  float sc[16];
  float mx = -1e30f;
  for (int t = 0; t < T; ++t) {
    float p = wave_sum(q * Kk[((size_t)t * B + b) * DM + h * 32 + lane]) * 0.17677669529663687f;
    if (causal && t > s) p = -1e9f;
    sc[t] = p; mx = fmaxf(mx, p);
  }
  float den = 0.f;
  for (int t = 0; t < T; ++t) { sc[t] = __expf(sc[t] - mx); den += sc[t]; }
  float o = 0.f;
  for (int t = 0; t < T; ++t) o += sc[t] * V[((size_t)t * B + b) * DM + h * 32 + lane];
  O[((size_t)s * B + b) * DM + h * 32 + lane] = o / den;
}

// final per-hour head dot: out[n, t] = relu_h1[t,n,:] . W2[t,:,0] + b2[t]
__global__ void head_final_k(const float* __restrict__ h1, const float* __restrict__ W2,
                             const float* __restrict__ b2, float* __restrict__ out) {
  int i = blockIdx.x * blockDim.x + threadIdx.x;
  if (i >= TP * NN) return;
  int t = i / NN, n = i % NN;
  const float* hp = h1 + ((size_t)t * NN + n) * 64;
  const float* wp = W2 + t * 64;
  float s = 0.f;
#pragma unroll
  for (int j = 0; j < 64; ++j) s += hp[j] * wp[j];
  out[(size_t)n * TP + t] = s + b2[t];
  // Cluster barrier encoding exercise: S_NOP when dispatch is not clustered
  // (ISA 00_overview §3.2: ClusterID==0 -> barrier ops become NOP).
  __builtin_amdgcn_s_cluster_barrier();
}

// ---------------------------------------------------------------------------
// Host orchestration
// ---------------------------------------------------------------------------
static inline void gemm(hipStream_t st, const float* A, const float* W, const float* bias,
                        float* C, int M, int N, int K, int flags) {
  if ((M % 32 == 0) && (N % 32 == 0) && (K % 32 == 0)) {
    dim3 g((N + 127) / 128, M / 32);
    wmma_gemm_fast_k<<<g, 128, 0, st>>>(A, W, bias, C, M, N, K, flags);
  } else {
    dim3 g((N + 15) / 16, (M + 15) / 16);
    wmma_gemm_k<<<g, 32, 0, st>>>(A, W, bias, C, M, N, K, flags);
  }
}
static inline void ln(hipStream_t st, const float* X, const float* Res, const float* g_,
                      const float* b_, float* Y, int M, int D) {
  ln_res_k<<<M, 32, 0, st>>>(X, Res, g_, b_, Y, M, D);
}
static inline void fillz(hipStream_t st, float* p, size_t n) {
  fill_k<<<(unsigned)((n + 255) / 256), 256, 0, st>>>(p, 0.f, n);
}

struct RgcnLayerP { const float *Wrel, *Wroot, *alpha, *bias, *bn_b, *bn_g; };
struct AttnP { const float *Wk, *Wo, *Wq, *Wv, *bk, *bo, *bq, *bv; };

extern "C" void kernel_launch(void* const* d_in, const int* in_sizes, int n_in,
                              void* d_out, int out_size, void* d_ws, size_t ws_size,
                              hipStream_t stream) {
  (void)in_sizes; (void)n_in; (void)out_size; (void)ws_size;
  auto F = [&](int i) { return (const float*)d_in[i]; };
  auto I = [&](int i) { return (const int*)d_in[i]; };

  const int* batch = I(0);
  const int* src   = I(1);
  const int* dst   = I(1) + NE;
  const int* etype = I(2);
  const float* envF = F(3);
  const float* timeF = F(146);
  const float* xall  = F(147);

  // ---- workspace bump allocator (256B aligned chunks) ----
  size_t off = 0;
  auto alloc = [&](size_t n) {
    float* p = (float*)((char*)d_ws + off);
    off += ((n * sizeof(float) + 255) / 256) * 256;
    return p;
  };
  float* CNT  = alloc(NN * NREL);
  float* INV  = alloc(NN * NREL);
  float* ENVH = alloc(13 * NG * 16);
  float* ENVO = alloc(13 * NG * 16);
  float* TIMH = alloc(TP * 7);
  float* TIMO = alloc(TP * 8);
  float* Z6   = alloc(8);
  float* ETH  = alloc(8);
  float* ETO  = alloc(8);
  float* PE   = alloc(TP * DM);
  // encoder stream (8000 rows)
  float* EX32 = alloc((size_t)NN * 32);
  float* EXA  = alloc((size_t)NN * DM);
  float* EXB  = alloc((size_t)NN * DM);
  float* EXW  = alloc((size_t)NN * DM);
  float* EAGG = alloc((size_t)NN * DM);
  float* EFIN = alloc((size_t)NN * FIN);
  float* EFH  = alloc((size_t)NN * FHID);
  float* EQ   = alloc((size_t)NN * DM);
  float* EK   = alloc((size_t)NN * DM);
  float* EV   = alloc((size_t)NN * DM);
  float* EO   = alloc((size_t)NN * DM);
  float* EP   = alloc((size_t)NN * DM);
  float* EFF  = alloc((size_t)NN * DFF);
  float* MEM  = alloc((size_t)NN * DM);
  // decoder stream (96000 rows) -- heavily reused
  float* DX32 = alloc((size_t)ND * 32);
  float* DXA  = alloc((size_t)ND * DM);   // also attn V
  float* DXB  = alloc((size_t)ND * DM);   // becomes tgt
  float* DXW  = alloc((size_t)ND * DM);   // also attn Q, head hidden
  float* DAGG = alloc((size_t)ND * DM);   // also attn K
  float* DFIN = alloc((size_t)ND * FIN);  // also attn O
  float* DFH  = alloc((size_t)ND * FHID); // also residual branch P

  // ---- edge degree (shared by all RGCN layers) ----
  fillz(stream, CNT, NN * NREL);
  edge_cnt_k<<<(NE + 255) / 256, 256, 0, stream>>>(dst, etype, CNT, NE);
  inv_cnt_k<<<(NN * NREL + 255) / 256, 256, 0, stream>>>(CNT, INV, NN * NREL);

  // ---- small MLP encoders ----
  auto run_mlp = [&](const float* X, int M, int Kin, int H, int Dout, int pb,
                     float* HID, float* OUT) {
    gemm(stream, X, F(pb + 0), F(pb + 2), HID, M, H, Kin, GF_RELU);
    ln(stream, HID, nullptr, F(pb + 5), F(pb + 4), HID, M, H);
    gemm(stream, HID, F(pb + 1), F(pb + 3), OUT, M, Dout, H, 0);
  };
  run_mlp(envF, 13 * NG, 16, 16, 16, 88, ENVH, ENVO);   // env encoder
  run_mlp(timeF, TP, 6, 7, 8, 140, TIMH, TIMO);         // time encoder (12 steps)
  fillz(stream, Z6, 6);
  run_mlp(Z6, 1, 6, 7, 8, 140, ETH, ETO);               // encoder-step time emb

  compute_pe_k<<<(TP * DM + 255) / 256, 256, 0, stream>>>(PE, TP);

  // ---- RGCN block runner ----
  auto run_rgcn = [&](const float* X32, float* XA, float* XB, float* XW, float* AGG,
                      int rgcn_base, int S) {
    int ROWS = S * NN;
    const float* cur = X32;
    int Kin = 32;
    float* outBuf[3] = {XA, XB, XA};
    for (int l = 0; l < 3; ++l) {
      RgcnLayerP L = {F(rgcn_base + l * 6 + 0), F(rgcn_base + l * 6 + 1),
                      F(rgcn_base + l * 6 + 2), F(rgcn_base + l * 6 + 3),
                      F(rgcn_base + l * 6 + 4), F(rgcn_base + l * 6 + 5)};
      size_t an = (size_t)ROWS * DM;
      fillz(stream, AGG, an);
      for (int r = 0; r < NREL; ++r) {
        gemm(stream, cur, L.Wrel + (size_t)r * Kin * DM, nullptr, XW, ROWS, DM, Kin, 0);
        int tot = NE * S;
        rgcn_scatter_k<<<(tot + 255) / 256, 256, 0, stream>>>(XW, src, dst, etype, INV,
                                                              AGG, r, NE, S, DM);
      }
      gemm(stream, cur, L.Wroot, L.bias, AGG, ROWS, DM, Kin, GF_ACC);
      bn_prelu_k<<<(unsigned)((an + 255) / 256), 256, 0, stream>>>(AGG, L.bn_g, L.bn_b,
                                                                   L.alpha, outBuf[l], an, DM);
      cur = outBuf[l];
      Kin = DM;
    }
  };

  auto attnP = [&](int base) {
    AttnP a = {F(base + 0), F(base + 1), F(base + 2), F(base + 3),
               F(base + 4), F(base + 5), F(base + 6), F(base + 7)};
    return a;
  };

  const float XSCL = 1.0f / (1.0f + 1e-8f);

  // =================== ENCODER (t = 0) ===================
  scale_copy_k<<<((size_t)NN * 32 + 255) / 256, 256, 0, stream>>>(xall, EX32, XSCL,
                                                                  (size_t)NN * 32);
  run_rgcn(EX32, EXA, EXB, EXW, EAGG, 122, 1);          // rgcn_enc; result in EXA
  {
    size_t tot = (size_t)NN * FIN;
    fusion_concat_k<<<(unsigned)((tot + 255) / 256), 256, 0, stream>>>(EXA, ENVO, ETO, batch,
                                                                       EFIN, 1, 0, 0);
    gemm(stream, EFIN, F(94), F(96), EFH, NN, FHID, FIN, GF_RELU);
    ln(stream, EFH, nullptr, F(99), F(98), EFH, NN, FHID);
    gemm(stream, EFH, F(95), F(97), MEM, NN, DM, FHID, 0);
    add_pe_k<<<((size_t)NN * DM + 255) / 256, 256, 0, stream>>>(MEM, PE, NN, (size_t)NN * DM);
  }
  for (int l = 0; l < 2; ++l) {
    int eb = 56 + l * 16;
    AttnP a = attnP(eb + 2);
    gemm(stream, MEM, a.Wq, a.bq, EQ, NN, DM, DM, 0);
    gemm(stream, MEM, a.Wk, a.bk, EK, NN, DM, DM, 0);
    gemm(stream, MEM, a.Wv, a.bv, EV, NN, DM, DM, 0);
    mha_attn_k<<<NN * 4, 32, 0, stream>>>(EQ, EK, EV, EO, 1, 1, NN, 0);
    gemm(stream, EO, a.Wo, a.bo, EP, NN, DM, DM, 0);
    ln(stream, EP, MEM, F(eb + 13), F(eb + 12), MEM, NN, DM);       // n1
    gemm(stream, MEM, F(eb + 0), F(eb + 10), EFF, NN, DFF, DM, GF_RELU);
    gemm(stream, EFF, F(eb + 1), F(eb + 11), EP, NN, DM, DFF, 0);
    ln(stream, EP, MEM, F(eb + 15), F(eb + 14), MEM, NN, DM);       // n2
  }

  // =================== DECODER (t = 1..12) ===================
  scale_copy_k<<<((size_t)ND * 32 + 255) / 256, 256, 0, stream>>>(xall + (size_t)NN * 32,
                                                                  DX32, XSCL, (size_t)ND * 32);
  run_rgcn(DX32, DXA, DXB, DXW, DAGG, 104, TP);         // rgcn_dec; result in DXA
  float* TGT = DXB;
  {
    size_t tot = (size_t)ND * FIN;
    fusion_concat_k<<<(unsigned)((tot + 255) / 256), 256, 0, stream>>>(DXA, ENVO, TIMO, batch,
                                                                       DFIN, TP, 1, 8);
    gemm(stream, DFIN, F(94), F(96), DFH, ND, FHID, FIN, GF_RELU);
    ln(stream, DFH, nullptr, F(99), F(98), DFH, ND, FHID);
    gemm(stream, DFH, F(95), F(97), TGT, ND, DM, FHID, 0);
    add_pe_k<<<((size_t)ND * DM + 255) / 256, 256, 0, stream>>>(TGT, PE, NN, (size_t)ND * DM);
  }
  float* Qb = DXW; float* Kb = DAGG; float* Vb = DXA; float* Ob = DFIN; float* Pb = DFH;
  for (int l = 0; l < 2; ++l) {
    int db = 4 + l * 26;
    AttnP sf = attnP(db + 18);   // self
    AttnP cr = attnP(db + 4);    // cross
    // masked self-attention
    gemm(stream, TGT, sf.Wq, sf.bq, Qb, ND, DM, DM, 0);
    gemm(stream, TGT, sf.Wk, sf.bk, Kb, ND, DM, DM, 0);
    gemm(stream, TGT, sf.Wv, sf.bv, Vb, ND, DM, DM, 0);
    mha_attn_k<<<TP * NN * 4, 32, 0, stream>>>(Qb, Kb, Vb, Ob, TP, TP, NN, 1);
    gemm(stream, Ob, sf.Wo, sf.bo, Pb, ND, DM, DM, 0);
    ln(stream, Pb, TGT, F(db + 13), F(db + 12), TGT, ND, DM);       // n1
    // cross-attention vs encoder memory (T=1)
    gemm(stream, TGT, cr.Wq, cr.bq, Qb, ND, DM, DM, 0);
    gemm(stream, MEM, cr.Wk, cr.bk, EK, NN, DM, DM, 0);
    gemm(stream, MEM, cr.Wv, cr.bv, EV, NN, DM, DM, 0);
    mha_attn_k<<<TP * NN * 4, 32, 0, stream>>>(Qb, EK, EV, Ob, TP, 1, NN, 0);
    gemm(stream, Ob, cr.Wo, cr.bo, Pb, ND, DM, DM, 0);
    ln(stream, Pb, TGT, F(db + 15), F(db + 14), TGT, ND, DM);       // n2
    // feed-forward (chunked per step to reuse the small 8000x512 buffer)
    for (int s = 0; s < TP; ++s) {
      gemm(stream, TGT + (size_t)s * NN * DM, F(db + 0), F(db + 2), EFF, NN, DFF, DM, GF_RELU);
      gemm(stream, EFF, F(db + 1), F(db + 3), Pb + (size_t)s * NN * DM, NN, DM, DFF, 0);
    }
    ln(stream, Pb, TGT, F(db + 17), F(db + 16), TGT, ND, DM);       // n3
  }

  // =================== per-hour heads ===================
  float* H1 = DXW;
  for (int t = 0; t < TP; ++t)
    gemm(stream, TGT + (size_t)t * NN * DM, F(100) + (size_t)t * DM * 64, F(102) + t * 64,
         H1 + (size_t)t * NN * 64, NN, 64, DM, GF_RELU);
  head_final_k<<<(TP * NN + 255) / 256, 256, 0, stream>>>(H1, F(101), F(103), (float*)d_out);
}
